// FastFFTNet_69312182223120
// MI455X (gfx1250) — matrix operand order repaired and checked by reading, not compiled, not run on hardware
//
#include <hip/hip_runtime.h>
#include <cstddef>

#define N_LAYERS 11
#define FFT      256
#define NUM_CLS  256
#define COND     80
#define KPAD     96
#define FRAMES   16
#define HOP      128
#define T_TOTAL  (FRAMES * HOP)          // 2048
#define CONDROWS (N_LAYERS * FFT)        // 2816
#define RING_F   (2047 * FFT)            // sum(2^j) cols * 256 ch
#define WELEM    (N_LAYERS * FFT * FFT)  // 720896 per weight tensor

typedef _Float16 v16h __attribute__((ext_vector_type(16)));
typedef float    v8f  __attribute__((ext_vector_type(8)));
typedef _Float16 h8v  __attribute__((ext_vector_type(8)));

// ---------------------------------------------------------------------------
// Prep kernels: tanh(emb), zero ring, f32->f16 weight conversion,
// zero-padded f16 A [2816x96], n-major f16 B [2048x96] (B[t][k] = y[k][t>>7]).
// ---------------------------------------------------------------------------
__global__ void emb_tanh_kernel(const float* __restrict__ er, float* __restrict__ emb) {
    int i = blockIdx.x * blockDim.x + threadIdx.x;
    if (i < NUM_CLS * FFT) emb[i] = tanhf(er[i]);
}

__global__ void zero_f32_kernel(float* __restrict__ p, int n) {
    int i = blockIdx.x * blockDim.x + threadIdx.x;
    if (i < n) p[i] = 0.0f;
}

__global__ void cvt_f16_kernel(const float* __restrict__ s, _Float16* __restrict__ d, int n) {
    int i = blockIdx.x * blockDim.x + threadIdx.x;
    if (i < n) d[i] = (_Float16)s[i];
}

__global__ void pack_A_kernel(const float* __restrict__ Wc, _Float16* __restrict__ Apad) {
    int i = blockIdx.x * blockDim.x + threadIdx.x;
    if (i >= CONDROWS * KPAD) return;
    int r = i / KPAD, k = i % KPAD;
    Apad[i] = (k < COND) ? (_Float16)Wc[r * COND + k] : (_Float16)0.0f;
}

__global__ void pack_B_kernel(const float* __restrict__ y, _Float16* __restrict__ Bpad) {
    int i = blockIdx.x * blockDim.x + threadIdx.x;
    if (i >= T_TOTAL * KPAD) return;
    int t = i / KPAD, k = i % KPAD;
    Bpad[i] = (k < COND) ? (_Float16)y[k * FRAMES + (t >> 7)] : (_Float16)0.0f;
}

// ---------------------------------------------------------------------------
// conds[t][2816] = (condition_W @ yu)^T, one 16x16 tile per wave.
// Branch-free fragments: two b128 loads each for A and B per K-step.
// ---------------------------------------------------------------------------
__global__ void cond_gemm_wmma(const _Float16* __restrict__ Apad,
                               const _Float16* __restrict__ Bpad,
                               float* __restrict__ conds) {
    const int lane   = threadIdx.x & 31;
    const int wid    = blockIdx.x * (blockDim.x >> 5) + (threadIdx.x >> 5);
    const int tilesN = T_TOTAL / 16;                 // 128
    const int tilesM = CONDROWS / 16;                // 176
    if (wid >= tilesM * tilesN) return;
    const int tM = wid / tilesN;
    const int tN = wid % tilesN;

    const int mrow  = tM * 16 + (lane & 15);
    const int ncol  = tN * 16 + (lane & 15);
    const int abase = (lane >= 16) ? 8  : 0;         // 16-bit A 16x32 layout
    const int bbase = (lane >= 16) ? 8  : 0;         // 16-bit B 32x16 layout
    const _Float16* ap = Apad + (size_t)mrow * KPAD;
    const _Float16* bp = Bpad + (size_t)ncol * KPAD;

    v8f acc = {};
#pragma unroll
    for (int ks = 0; ks < KPAD; ks += 32) {
        h8v alo = *reinterpret_cast<const h8v*>(ap + ks + abase);
        h8v ahi = *reinterpret_cast<const h8v*>(ap + ks + 16 + abase);
        h8v blo = *reinterpret_cast<const h8v*>(bp + ks + bbase);
        h8v bhi = *reinterpret_cast<const h8v*>(bp + ks + 16 + bbase);
        v16h a = __builtin_shufflevector(alo, ahi, 0,1,2,3,4,5,6,7,8,9,10,11,12,13,14,15);
        v16h b = __builtin_shufflevector(blo, bhi, 0,1,2,3,4,5,6,7,8,9,10,11,12,13,14,15);
        acc = __builtin_amdgcn_wmma_f32_16x16x32_f16(
            false, a, false, b, (short)0, acc, false, false);
    }

    const int colt  = tN * 16 + (lane & 15);
    const int rbase = tM * 16 + ((lane >= 16) ? 8 : 0);
#pragma unroll
    for (int r = 0; r < 8; ++r)                      // t-major store
        conds[(size_t)colt * CONDROWS + (rbase + r)] = acc[r];
}

// ---------------------------------------------------------------------------
// Persistent sequential decode: 1 WGP, 1024 threads (32 wave32).
// f16 weights with f32 accumulate; 4 lanes per output row; shfl reduce.
// ---------------------------------------------------------------------------
__device__ __forceinline__ float dot64h(const _Float16* __restrict__ w, const float* v) {
    const h8v*    w8 = reinterpret_cast<const h8v*>(w);
    const float4* v4 = reinterpret_cast<const float4*>(v);
    float s = 0.0f;
#pragma unroll
    for (int i = 0; i < 8; ++i) {
        h8v a = w8[i];
        float4 b0 = v4[2 * i];
        float4 b1 = v4[2 * i + 1];
        s = fmaf((float)a[0], b0.x, s);
        s = fmaf((float)a[1], b0.y, s);
        s = fmaf((float)a[2], b0.z, s);
        s = fmaf((float)a[3], b0.w, s);
        s = fmaf((float)a[4], b1.x, s);
        s = fmaf((float)a[5], b1.y, s);
        s = fmaf((float)a[6], b1.z, s);
        s = fmaf((float)a[7], b1.w, s);
    }
    return s;
}

__global__ __launch_bounds__(1024)
void decode_kernel(const float* __restrict__ conds,
                   const float* __restrict__ samples,
                   const float* __restrict__ emb,
                   const _Float16* __restrict__ WVp,
                   const _Float16* __restrict__ WVpr,
                   const _Float16* __restrict__ Wo,
                   const float* __restrict__ Wob,
                   const _Float16* __restrict__ ew,
                   const float* __restrict__ eb,
                   float* __restrict__ ring,
                   int* __restrict__ out) {
    __shared__ __align__(16) float xv[FFT];
    __shared__ __align__(16) float hv[FFT];
    __shared__ __align__(16) float mc[FFT];
    __shared__ __align__(16) float lg[FFT];
    __shared__ int sel;

    const int tid  = threadIdx.x;
    const int row  = tid >> 2;
    const int seg  = tid & 3;
    const int lane = tid & 31;

    if (tid < FFT) xv[tid] = emb[(NUM_CLS / 2 - 1) * FFT + tid];

    // software-pipelined ring column: nxt holds layer-0 column of t=0 (zeros)
    float nxt = (tid < FFT) ? ring[tid] : 0.0f;
    __syncthreads();

    for (int t = 0; t < T_TOTAL; ++t) {
        if (tid < 88 && t + 1 < T_TOTAL)             // prefetch next step's conds row
            __builtin_prefetch(conds + (size_t)(t + 1) * CONDROWS + tid * 32, 0, 1);

        for (int j = 0; j < N_LAYERS; ++j) {
            const int m   = 1 << j;
            const int idx = t & (m - 1);
            float* rc = ring + (size_t)(m - 1) * FFT + (size_t)idx * FFT;

            if (tid < FFT) mc[tid] = nxt;            // consume pipelined column
            __syncthreads();

            // issue load of the NEXT layer's column (next step's layer 0 at j==10);
            // its writer runs only after this value is consumed -> race-free
            {
                const int jn  = (j + 1 < N_LAYERS) ? j + 1 : 0;
                const int tn  = (j + 1 < N_LAYERS) ? t : t + 1;
                const int mn  = 1 << jn;
                const int in_ = tn & (mn - 1);
                const float* rcn = ring + (size_t)(mn - 1) * FFT + (size_t)in_ * FFT;
                if (tid < FFT) nxt = rcn[tid];
            }

            // write current x into this layer's ring slot (column already staged)
            if (tid < FFT) rc[tid] = xv[tid];

            const size_t wbase = (size_t)j * FFT * FFT + (size_t)row * FFT + seg * 64;
            __builtin_prefetch(Wo + wbase, 0, 1);    // global_prefetch_b8
            float acc = dot64h(WVp  + wbase, mc + seg * 64)
                      + dot64h(WVpr + wbase, xv + seg * 64);
            acc += __shfl_xor(acc, 1, 32);
            acc += __shfl_xor(acc, 2, 32);
            if (seg == 0) {
                float c = conds[(size_t)t * CONDROWS + j * FFT + row];
                hv[row] = fmaxf(c + acc, 0.0f);      // relu(h)
            }
            __syncthreads();

            float a2 = dot64h(Wo + wbase, hv + seg * 64);
            a2 += __shfl_xor(a2, 1, 32);
            a2 += __shfl_xor(a2, 2, 32);
            if (seg == 0) {
                float z = a2 + Wob[j * FFT + row];
                xv[row] = fmaxf(z + xv[row], 0.0f);  // relu(z + x)
            }
            __syncthreads();
        }

        // logits = end_w @ x + end_b
        {
            const size_t wbase = (size_t)row * FFT + seg * 64;
            float a3 = dot64h(ew + wbase, xv + seg * 64);
            a3 += __shfl_xor(a3, 1, 32);
            a3 += __shfl_xor(a3, 2, 32);
            if (seg == 0) lg[row] = a3 + eb[row];
        }
        __syncthreads();

        // softmax CDF inversion: first i with cumsum(e) > u * sum(e)
        if (tid < 32) {
            float e[8];
            float mx = -1e30f;
#pragma unroll
            for (int i = 0; i < 8; ++i) mx = fmaxf(mx, lg[lane * 8 + i]);
            for (int o = 16; o >= 1; o >>= 1) mx = fmaxf(mx, __shfl_xor(mx, o, 32));

            float cs = 0.0f;
#pragma unroll
            for (int i = 0; i < 8; ++i) { e[i] = expf(lg[lane * 8 + i] - mx); cs += e[i]; }

            float inc = cs;                           // inclusive scan over lanes
            for (int o = 1; o < 32; o <<= 1) {
                float nb = __shfl_up(inc, o, 32);
                if (lane >= o) inc += nb;
            }
            float total = __shfl(inc, 31, 32);
            float thr   = samples[t] * total;

            float run = inc - cs;                     // exclusive prefix
            int found = 1 << 30;
#pragma unroll
            for (int i = 0; i < 8; ++i) {
                run += e[i];
                if (run > thr && found == (1 << 30)) found = lane * 8 + i;
            }
            for (int o = 16; o >= 1; o >>= 1) {
                int other = __shfl_xor(found, o, 32);
                found = (other < found) ? other : found;
            }
            if (lane == 0) {
                int s = (found >= NUM_CLS) ? 0 : found;   // jnp.argmax fallback
                sel = s;
                out[t] = s;
            }
        }
        __syncthreads();
        if (tid < FFT) xv[tid] = emb[(size_t)sel * FFT + tid];
        __syncthreads();
    }
}

// ---------------------------------------------------------------------------
extern "C" void kernel_launch(void* const* d_in, const int* in_sizes, int n_in,
                              void* d_out, int out_size, void* d_ws, size_t ws_size,
                              hipStream_t stream) {
    (void)in_sizes; (void)n_in; (void)out_size; (void)ws_size;
    const float* y    = (const float*)d_in[0];
    const float* smp  = (const float*)d_in[1];
    const float* embr = (const float*)d_in[2];
    const float* Wc   = (const float*)d_in[3];
    const float* WVp  = (const float*)d_in[4];
    const float* WVpr = (const float*)d_in[5];
    const float* Wo   = (const float*)d_in[6];
    const float* Wob  = (const float*)d_in[7];
    const float* ew   = (const float*)d_in[8];
    const float* eb   = (const float*)d_in[9];
    int* out = (int*)d_out;

    char* ws = (char*)d_ws;
    size_t off = 0;
    float* conds = (float*)(ws + off); off += (size_t)T_TOTAL * CONDROWS * 4;   // t-major
    float* emb   = (float*)(ws + off); off += (size_t)NUM_CLS * FFT * 4;
    float* ring  = (float*)(ws + off); off += (size_t)RING_F * 4;
    _Float16* Apad  = (_Float16*)(ws + off); off += (size_t)CONDROWS * KPAD * 2;
    _Float16* Bpad  = (_Float16*)(ws + off); off += (size_t)T_TOTAL * KPAD * 2;
    _Float16* WVp16 = (_Float16*)(ws + off); off += (size_t)WELEM * 2;
    _Float16* WVpr16= (_Float16*)(ws + off); off += (size_t)WELEM * 2;
    _Float16* Wo16  = (_Float16*)(ws + off); off += (size_t)WELEM * 2;
    _Float16* ew16  = (_Float16*)(ws + off); off += (size_t)NUM_CLS * FFT * 2;

    emb_tanh_kernel<<<(NUM_CLS * FFT + 255) / 256, 256, 0, stream>>>(embr, emb);
    pack_A_kernel<<<(CONDROWS * KPAD + 255) / 256, 256, 0, stream>>>(Wc, Apad);
    pack_B_kernel<<<(T_TOTAL * KPAD + 255) / 256, 256, 0, stream>>>(y, Bpad);
    cvt_f16_kernel<<<(WELEM + 255) / 256, 256, 0, stream>>>(WVp,  WVp16,  WELEM);
    cvt_f16_kernel<<<(WELEM + 255) / 256, 256, 0, stream>>>(WVpr, WVpr16, WELEM);
    cvt_f16_kernel<<<(WELEM + 255) / 256, 256, 0, stream>>>(Wo,   Wo16,   WELEM);
    cvt_f16_kernel<<<(NUM_CLS * FFT + 255) / 256, 256, 0, stream>>>(ew, ew16, NUM_CLS * FFT);
    zero_f32_kernel<<<(RING_F + 255) / 256, 256, 0, stream>>>(ring, RING_F);

    const int nwaves = (CONDROWS / 16) * (T_TOTAL / 16);
    cond_gemm_wmma<<<(nwaves + 7) / 8, 256, 0, stream>>>(Apad, Bpad, conds);

    decode_kernel<<<1, 1024, 0, stream>>>(conds, smp, emb, WVp16, WVpr16,
                                          Wo16, Wob, ew16, eb, ring, out);
}
